// InteractionBlock_18081812316197
// MI455X (gfx1250) — matrix operand order repaired and checked by reading, not compiled, not run on hardware
//
#include <hip/hip_runtime.h>

#define HID 128
#define NG  50

typedef __attribute__((ext_vector_type(16))) __bf16 v16bf;
typedef __attribute__((ext_vector_type(8)))  float  v8f;
typedef __attribute__((ext_vector_type(4)))  unsigned u32x4;
typedef __attribute__((ext_vector_type(8)))  int      i32x8;
typedef __attribute__((ext_vector_type(4)))  int      i32x4;

union FragA {
  v16bf bf;
  unsigned u[8];
};

// fp32 -> bf16 with round-to-nearest-even
__device__ __forceinline__ unsigned short f2bf(float x) {
  unsigned u = __float_as_uint(x);
  unsigned r = u + 0x7FFFu + ((u >> 16) & 1u);
  return (unsigned short)(r >> 16);
}
__device__ __forceinline__ unsigned pack2(float lo, float hi) {
  return (unsigned)f2bf(lo) | ((unsigned)f2bf(hi) << 16);
}

// A-matrix 16x32 bf16 layout (ISA 7.12.2): lane l holds row m=l&15;
// VGPR v covers K pair kbase = (v>>2)*16 + ((l>>4)&1)*8 + (v&3)*2.
// Returned value is the *pair* index (kbase/2) into a packed row.
__device__ __forceinline__ int a_kp(int v, int lane) {
  return ((v >> 2) << 3) + (((lane >> 4) & 1) << 2) + (v & 3);
}
// B-matrix 32x16 bf16: same K order as A; VGPR v covers K pair
// (v>>2)*16 + ((l>>4)&1)*8 + (v&3)*2.
__device__ __forceinline__ int b_kp(int v, int lane) {
  return ((v >> 2) << 3) + (((lane >> 4) & 1) << 2) + (v & 3);
}

__device__ __forceinline__ FragA load_a(const unsigned* row, int lane, int po) {
  FragA f;
#pragma unroll
  for (int v = 0; v < 8; ++v) f.u[v] = row[po + a_kp(v, lane)];
  return f;
}
__device__ __forceinline__ FragA load_b(const unsigned* row, int lane, int po) {
  FragA f;
#pragma unroll
  for (int v = 0; v < 8; ++v) f.u[v] = row[po + b_kp(v, lane)];
  return f;
}

__device__ __forceinline__ v8f wmma_bf16(FragA a, FragA b, v8f c) {
  return __builtin_amdgcn_wmma_f32_16x16x32_bf16(
      /*neg_a=*/false, a.bf, /*neg_b=*/false, b.bf,
      /*c_mod=*/(short)0, c, /*reuse_a=*/false, /*reuse_b=*/false);
}

__device__ __forceinline__ float silu_f(float x) { return x / (1.0f + __expf(-x)); }

// ---------------------------------------------------------------------------
// Tensor Data Mover: DMA a rows x cols fp32 tile (row stride in elements)
// from global memory into contiguous LDS at lds_addr. D# per ISA ch.8.
// This toolchain exposes the 6-arg builtin (g0, g1, g2, g3, aux, cpol).
// ---------------------------------------------------------------------------
__device__ __forceinline__ void tdm_load_tile_f32(unsigned lds_addr,
                                                  const float* gptr,
                                                  unsigned rows, unsigned cols,
                                                  unsigned stride) {
  unsigned long long ga = (unsigned long long)(uintptr_t)gptr;
  // group0: count=1 | lds_addr | global_addr[56:0] | type=2 ("image")
  u32x4 g0 = {1u, lds_addr, (unsigned)(ga & 0xFFFFFFFFull),
              (unsigned)((ga >> 32) & 0x1FFFFFFull) | (2u << 30)};
  // group1: wg_mask=0, data_size=2 (4B), no barrier/iterate/pad;
  //   tensor_dim0=cols, tensor_dim1=rows, tile_dim0=cols, tile_dim1=rows,
  //   tile_dim2=0, tensor_dim0_stride=stride, tensor_dim1_stride=0
  i32x8 g1 = {(int)(2u << 16),
              (int)(cols << 16),           // tensor_dim0[15:0] @ bits 63:48
              (int)(rows << 16),           // dim0 hi=0 | tensor_dim1[15:0]
              (int)(cols << 16),           // dim1 hi=0 | tile_dim0
              (int)rows,                   // tile_dim1 | tile_dim2=0
              (int)stride,                 // tensor_dim0_stride[31:0]
              0, 0};
  // group2: tensor_dim2=1, tensor_dim3=0, dim2_stride=0, tile_dim3=0
  i32x4 g2 = {1, 0, 0, 0};
  i32x4 g3 = {0, 0, 0, 0};
  i32x8 aux = {0, 0, 0, 0, 0, 0, 0, 0};
  __builtin_amdgcn_tensor_load_to_lds(g0, g1, g2, g3, aux, 0);
}

// ---------------------------------------------------------------------------
// Zero the scatter target in workspace
// ---------------------------------------------------------------------------
__global__ void k_zero(float4* __restrict__ p, int n4) {
  int i = blockIdx.x * blockDim.x + threadIdx.x;
  if (i < n4) p[i] = make_float4(0.f, 0.f, 0.f, 0.f);
}

// ---------------------------------------------------------------------------
// xf = x @ lin1_w^T        [N,128] = [N,128] x [128,128]^T
// block = 128 threads (4 waves), each wave computes a 16-row tile
// ---------------------------------------------------------------------------
__global__ __launch_bounds__(128) void k_xf(const float* __restrict__ x,
                                            const float* __restrict__ w,
                                            float* __restrict__ xf, int nrows) {
  __shared__ unsigned wp[128 * 64];                 // lin1_w packed bf16 pairs
  __shared__ unsigned short tilebuf[4][16 * 128];   // per-wave x tile (bf16)
  const int tid = threadIdx.x, lane = tid & 31, wv = tid >> 5;

  {  // stage weights: thread t owns row t
    const float* wr = w + tid * HID;
    unsigned* dp = wp + tid * 64;
#pragma unroll 8
    for (int p = 0; p < 64; ++p) dp[p] = pack2(wr[2 * p], wr[2 * p + 1]);
  }
  __syncthreads();

  const int r0 = (blockIdx.x * 4 + wv) * 16;
  const bool valid = r0 < nrows;
  const int r0c = valid ? r0 : 0;

  unsigned* tp = (unsigned*)&tilebuf[wv][0];        // [16][64] packed pairs
#pragma unroll 4
  for (int i = 0; i < 32; ++i) {                    // 1024 packs / 32 lanes
    int idx = i * 32 + lane;
    int m = idx >> 6, p = idx & 63;
    const float* xr = x + (long)(r0c + m) * HID;
    tp[m * 64 + p] = pack2(xr[2 * p], xr[2 * p + 1]);
  }
  __syncthreads();

  FragA a[4];
  const unsigned* arow = tp + (lane & 15) * 64;
#pragma unroll
  for (int ks = 0; ks < 4; ++ks) a[ks] = load_a(arow, lane, ks * 16);

  const int n0 = lane & 15, mb = (lane >> 4) * 8;
#pragma unroll
  for (int nt = 0; nt < 8; ++nt) {
    const int n = nt * 16 + n0;
    const unsigned* brow = wp + n * 64;
    v8f acc = {0.f, 0.f, 0.f, 0.f, 0.f, 0.f, 0.f, 0.f};
#pragma unroll
    for (int ks = 0; ks < 4; ++ks) acc = wmma_bf16(a[ks], load_b(brow, lane, ks * 16), acc);
    if (valid) {
#pragma unroll
      for (int r = 0; r < 8; ++r) xf[(long)(r0 + mb + r) * HID + n] = acc[r];
    }
  }
}

// ---------------------------------------------------------------------------
// Fused edge kernel:
//   h  = silu(attr @ w1^T + b1)            (K=50 padded to 64)
//   W  = (h @ w2^T + b2) * cutoff(e)
//   atomically: agg[dst] += xf[src] * W
// block = 128 threads (4 waves), 16 edges per wave.
// The 16x50 fp32 attr tile is DMA'd into LDS by the Tensor Data Mover,
// issued before weight staging so HBM latency overlaps the bf16 packing.
// ---------------------------------------------------------------------------
__global__ __launch_bounds__(128) void k_edge(
    const float* __restrict__ attr, const float* __restrict__ w1,
    const float* __restrict__ b1, const float* __restrict__ w2,
    const float* __restrict__ b2, const long long* __restrict__ eidx,
    const float* __restrict__ ew, const float* __restrict__ xf,
    float* __restrict__ agg, int E) {
  __shared__ unsigned w1p[128 * 32];               // mlp_w1 packed (K pad 64)
  __shared__ unsigned w2p[128 * 64];               // mlp_w2 packed (K=128)
  __shared__ unsigned short hbuf[4][16 * 128];     // attr(bf16) then h tile
  __shared__ float rawattr[4][16 * NG];            // TDM destination (fp32)
  __shared__ float cbuf[4][16];
  __shared__ int   sbuf[4][16], dbuf[4][16];
  const int tid = threadIdx.x, lane = tid & 31, wv = tid >> 5;

  const int e0 = (blockIdx.x * 4 + wv) * 16;
  const bool valid = e0 < E;
  const int e0c = valid ? e0 : 0;

  // 1) kick off the async tensor DMA for this wave's attr tile
  tdm_load_tile_f32((unsigned)(uintptr_t)&rawattr[wv][0],
                    attr + (long)e0c * NG, 16u, (unsigned)NG, (unsigned)NG);

  // 2) stage both weight matrices (overlaps with the TDM transfer)
  {
    const float* r1 = w1 + tid * NG;
    unsigned* d1 = w1p + tid * 32;
#pragma unroll 8
    for (int p = 0; p < 32; ++p) {
      float lo = (2 * p     < NG) ? r1[2 * p]     : 0.f;
      float hi = (2 * p + 1 < NG) ? r1[2 * p + 1] : 0.f;
      d1[p] = pack2(lo, hi);
    }
    const float* r2 = w2 + tid * HID;
    unsigned* d2 = w2p + tid * 64;
#pragma unroll 8
    for (int p = 0; p < 64; ++p) d2[p] = pack2(r2[2 * p], r2[2 * p + 1]);
  }
  if (lane < 16) {
    int e = e0c + lane;
    cbuf[wv][lane] = 0.5f * (__cosf(ew[e] * 0.31415926535f) + 1.0f);
    int s = (int)eidx[e];
    sbuf[wv][lane] = s;
    dbuf[wv][lane] = (int)eidx[(long)E + e];
    __builtin_prefetch(&xf[(long)s * HID], 0, 3);  // warm gather rows (L2)
  }
  __syncthreads();

  // 3) wait for this wave's tensor DMA, then convert fp32 -> packed bf16
  __builtin_amdgcn_s_wait_tensorcnt(0);
  unsigned* tp = (unsigned*)&hbuf[wv][0];          // row stride 64 uints
  const float* raw = &rawattr[wv][0];
#pragma unroll 4
  for (int i = 0; i < 16; ++i) {
    float lo = 0.f, hi = 0.f;
    if (lane < 25) {                               // 2*lane, 2*lane+1 < 50
      lo = raw[i * NG + 2 * lane];
      hi = raw[i * NG + 2 * lane + 1];
    }
    tp[i * 64 + lane] = pack2(lo, hi);             // K padded to 64
  }
  __syncthreads();

  // GEMM1: h = attr @ w1^T  (two K=32 steps)
  const unsigned* arow = tp + (lane & 15) * 64;
  FragA a0 = load_a(arow, lane, 0);
  FragA a1 = load_a(arow, lane, 16);
  const int n0 = lane & 15, mb = (lane >> 4) * 8;
  v8f hacc[8];
#pragma unroll
  for (int nt = 0; nt < 8; ++nt) {
    const int n = nt * 16 + n0;
    const unsigned* brow = w1p + n * 32;
    v8f acc = {0.f, 0.f, 0.f, 0.f, 0.f, 0.f, 0.f, 0.f};
    acc = wmma_bf16(a0, load_b(brow, lane, 0), acc);
    acc = wmma_bf16(a1, load_b(brow, lane, 16), acc);
    hacc[nt] = acc;
  }
  __syncthreads();

  // write silu(h + b1) back to LDS as bf16 (re-shape C-layout -> A-layout)
#pragma unroll
  for (int nt = 0; nt < 8; ++nt) {
    const int n = nt * 16 + n0;
    const float bias = b1[n];
#pragma unroll
    for (int r = 0; r < 8; ++r) {
      float hx = hacc[nt][r] + bias;
      hbuf[wv][(mb + r) * 128 + n] = f2bf(silu_f(hx));
    }
  }
  __syncthreads();

  // GEMM2: W = h @ w2^T (K=128), fold bias * cutoff, gather+scatter
  FragA a[4];
  const unsigned* hrow = tp + (lane & 15) * 64;
#pragma unroll
  for (int ks = 0; ks < 4; ++ks) a[ks] = load_a(hrow, lane, ks * 16);
#pragma unroll
  for (int nt = 0; nt < 8; ++nt) {
    const int n = nt * 16 + n0;
    const unsigned* brow = w2p + n * 64;
    v8f acc = {0.f, 0.f, 0.f, 0.f, 0.f, 0.f, 0.f, 0.f};
#pragma unroll
    for (int ks = 0; ks < 4; ++ks) acc = wmma_bf16(a[ks], load_b(brow, lane, ks * 16), acc);
    if (valid) {
      const float bias = b2[n];
#pragma unroll
      for (int r = 0; r < 8; ++r) {
        const int m = mb + r;
        float Wv = (acc[r] + bias) * cbuf[wv][m];
        float msg = xf[(long)sbuf[wv][m] * HID + n] * Wv;
        atomicAdd(&agg[(long)dbuf[wv][m] * HID + n], msg);
      }
    }
  }
}

// ---------------------------------------------------------------------------
// out = silu(agg @ lin2_w^T + lin2_b) @ lin_w^T + lin_b
// ---------------------------------------------------------------------------
__global__ __launch_bounds__(128) void k_out(
    const float* __restrict__ agg, const float* __restrict__ w2,
    const float* __restrict__ bias2, const float* __restrict__ w3,
    const float* __restrict__ bias3, float* __restrict__ out, int nrows) {
  __shared__ unsigned wp2[128 * 64], wp3[128 * 64];
  __shared__ unsigned short tbuf[4][16 * 128];
  const int tid = threadIdx.x, lane = tid & 31, wv = tid >> 5;

  {
    const float* r2 = w2 + tid * HID;
    const float* r3 = w3 + tid * HID;
    unsigned* d2 = wp2 + tid * 64;
    unsigned* d3 = wp3 + tid * 64;
#pragma unroll 8
    for (int p = 0; p < 64; ++p) {
      d2[p] = pack2(r2[2 * p], r2[2 * p + 1]);
      d3[p] = pack2(r3[2 * p], r3[2 * p + 1]);
    }
  }
  __syncthreads();

  const int r0 = (blockIdx.x * 4 + wv) * 16;
  const bool valid = r0 < nrows;
  const int r0c = valid ? r0 : 0;

  unsigned* tp = (unsigned*)&tbuf[wv][0];
#pragma unroll 4
  for (int i = 0; i < 32; ++i) {
    int idx = i * 32 + lane;
    int m = idx >> 6, p = idx & 63;
    const float* gr = agg + (long)(r0c + m) * HID;
    tp[m * 64 + p] = pack2(gr[2 * p], gr[2 * p + 1]);
  }
  __syncthreads();

  FragA a[4];
  const unsigned* arow = tp + (lane & 15) * 64;
#pragma unroll
  for (int ks = 0; ks < 4; ++ks) a[ks] = load_a(arow, lane, ks * 16);

  const int n0 = lane & 15, mb = (lane >> 4) * 8;
  v8f hacc[8];
#pragma unroll
  for (int nt = 0; nt < 8; ++nt) {
    const int n = nt * 16 + n0;
    const unsigned* brow = wp2 + n * 64;
    v8f acc = {0.f, 0.f, 0.f, 0.f, 0.f, 0.f, 0.f, 0.f};
#pragma unroll
    for (int ks = 0; ks < 4; ++ks) acc = wmma_bf16(a[ks], load_b(brow, lane, ks * 16), acc);
    hacc[nt] = acc;
  }
  __syncthreads();

#pragma unroll
  for (int nt = 0; nt < 8; ++nt) {
    const int n = nt * 16 + n0;
    const float bias = bias2[n];
#pragma unroll
    for (int r = 0; r < 8; ++r) {
      float hx = hacc[nt][r] + bias;
      tbuf[wv][(mb + r) * 128 + n] = f2bf(silu_f(hx));
    }
  }
  __syncthreads();

#pragma unroll
  for (int ks = 0; ks < 4; ++ks) a[ks] = load_a(arow, lane, ks * 16);
#pragma unroll
  for (int nt = 0; nt < 8; ++nt) {
    const int n = nt * 16 + n0;
    const unsigned* brow = wp3 + n * 64;
    v8f acc = {0.f, 0.f, 0.f, 0.f, 0.f, 0.f, 0.f, 0.f};
#pragma unroll
    for (int ks = 0; ks < 4; ++ks) acc = wmma_bf16(a[ks], load_b(brow, lane, ks * 16), acc);
    if (valid) {
      const float bias = bias3[n];
#pragma unroll
      for (int r = 0; r < 8; ++r) out[(long)(r0 + mb + r) * HID + n] = acc[r] + bias;
    }
  }
}

// ---------------------------------------------------------------------------
extern "C" void kernel_launch(void* const* d_in, const int* in_sizes, int n_in,
                              void* d_out, int out_size, void* d_ws, size_t ws_size,
                              hipStream_t stream) {
  const float* x        = (const float*)d_in[0];
  const long long* eidx = (const long long*)d_in[1];
  const float* ew       = (const float*)d_in[2];
  const float* attr     = (const float*)d_in[3];
  const float* mlp_w1   = (const float*)d_in[4];
  const float* mlp_b1   = (const float*)d_in[5];
  const float* mlp_w2   = (const float*)d_in[6];
  const float* mlp_b2   = (const float*)d_in[7];
  const float* lin1_w   = (const float*)d_in[8];
  const float* lin2_w   = (const float*)d_in[9];
  const float* lin2_b   = (const float*)d_in[10];
  const float* lin_w    = (const float*)d_in[11];
  const float* lin_b    = (const float*)d_in[12];

  const int N = in_sizes[0] / HID;   // 20000
  const int E = in_sizes[2];         // 640000

  float* xf  = (float*)d_ws;                      // [N,128]
  float* agg = xf + (size_t)N * HID;              // [N,128]

  const int total4 = (N * HID) / 4;
  k_zero<<<(total4 + 255) / 256, 256, 0, stream>>>((float4*)agg, total4);
  k_xf<<<(N + 63) / 64, 128, 0, stream>>>(x, lin1_w, xf, N);
  k_edge<<<(E + 63) / 64, 128, 0, stream>>>(attr, mlp_w1, mlp_b1, mlp_w2, mlp_b2,
                                            eidx, ew, xf, agg, E);
  k_out<<<(N + 63) / 64, 128, 0, stream>>>(agg, lin2_w, lin2_b, lin_w, lin_b,
                                           (float*)d_out, N);
}
